// NormGatedLinear_4638564680113
// MI455X (gfx1250) — matrix operand order repaired and checked by reading, not compiled, not run on hardware
//
#include <hip/hip_runtime.h>
#include <hip/hip_bf16.h>

#define HIDDEN      7168
#define NUM_EXPERTS 384
#define NUM_TOKENS  8192
#define RMS_EPS     1e-6f

typedef __attribute__((ext_vector_type(16))) _Float16 v16h;
typedef __attribute__((ext_vector_type(8)))  _Float16 v8h;
typedef __attribute__((ext_vector_type(8)))  float    v8f;

// ---------------------------------------------------------------------------
// Kernel 1: RMSNorm. One 256-thread block per token row (7168 = 256*28 floats).
// ---------------------------------------------------------------------------
__global__ __launch_bounds__(256)
void NormGatedLinear_rmsnorm_kernel(const float* __restrict__ x,
                                    const float* __restrict__ w,
                                    float* __restrict__ out) {
    const int row  = blockIdx.x;
    const int t    = threadIdx.x;
    const int lane = t & 31;
    const int wid  = t >> 5;

    const float4* xr = (const float4*)(x + (size_t)row * HIDDEN);
    float4*       yr = (float4*)(out + (size_t)row * HIDDEN);
    const float4* wr = (const float4*)w;

    float4 v[7];
    float ss = 0.f;
#pragma unroll
    for (int i = 0; i < 7; ++i) {
        v[i] = xr[t + i * 256];
        ss += v[i].x * v[i].x + v[i].y * v[i].y + v[i].z * v[i].z + v[i].w * v[i].w;
    }

#pragma unroll
    for (int off = 16; off > 0; off >>= 1) ss += __shfl_xor(ss, off);

    __shared__ float red[8];
    __shared__ float s_scale;
    if (lane == 0) red[wid] = ss;
    __syncthreads();
    if (wid == 0) {
        float r = (lane < 8) ? red[lane] : 0.f;
#pragma unroll
        for (int off = 4; off > 0; off >>= 1) r += __shfl_xor(r, off);
        if (lane == 0) s_scale = rsqrtf(r * (1.0f / (float)HIDDEN) + RMS_EPS);
    }
    __syncthreads();
    const float scale = s_scale;

#pragma unroll
    for (int i = 0; i < 7; ++i) {
        float4 wv = wr[t + i * 256];
        float4 o;
        o.x = v[i].x * scale * wv.x;
        o.y = v[i].y * scale * wv.y;
        o.z = v[i].z * scale * wv.z;
        o.w = v[i].w * scale * wv.w;
        yr[t + i * 256] = o;
    }
}

// ---------------------------------------------------------------------------
// Kernel 2: logits[T,E] = normed[T,H] @ gate_weight[E,H]^T via
// v_wmma_f32_16x16x32_f16 (f16 inputs, f32 accumulate).
//
// 128x128 workgroup tile, BK=32, 8 waves; wave w owns rows [w*16, w*16+16).
// Double-buffered LDS, software pipelined, branchless steady-state body
// (last K-step peeled), unroll-2 so the buffer index is static.
// B fragments loaded in two groups of 4 (32-VGPR peak instead of 64) so the
// allocator keeps acc[0..7] pinned in the WMMA D/C registers (no v_mov
// shuffles, no WAR-hazard nops); each group of 4 WMMAs still sees a clause
// of 8 ds_load_b128 ahead of it.
// ---------------------------------------------------------------------------
#define BM 128
#define BN 128
#define BK 32
#define LDSP 40   // f16 per LDS row: 32 data + 8 pad (80B pitch, bank spread)
#define NK (HIDDEN / BK)

__global__ __launch_bounds__(256, 1)
void NormGatedLinear_gate_gemm_kernel(const float* __restrict__ normed,
                                      const float* __restrict__ gateW,
                                      float* __restrict__ logits) {
    __shared__ _Float16 sA[2][BM * LDSP];
    __shared__ _Float16 sB[2][BN * LDSP];

    const int t    = threadIdx.x;
    const int lane = t & 31;
    const int wid  = t >> 5;
    // N fastest in launch order: consecutive blocks reuse the same A slab in L2
    const int n0   = blockIdx.x * BN;
    const int m0   = blockIdx.y * BM;

    // staging: 2 threads per row, 16 halves (4 float4) each
    const int srow = t >> 1;
    const int skc  = (t & 1) * 16;

    // fragment coords per ISA 16-bit layouts
    const int frow   = lane & 15;
    const int khalfA = (lane >> 4) * 8;   // A: two contiguous 8-half chunks
    const int khalfB = (lane >> 4) * 8;   // B: two contiguous 8-half chunks

    const float* aSrc = normed + (size_t)(m0 + srow) * HIDDEN + skc;
    const float* bSrc = gateW  + (size_t)(n0 + srow) * HIDDEN + skc;

    v8f acc[8];
#pragma unroll
    for (int i = 0; i < 8; ++i) acc[i] = (v8f){0.f, 0.f, 0.f, 0.f, 0.f, 0.f, 0.f, 0.f};

#define CVT_STORE(buf_, A0, A1, A2, A3, B0, B1, B2, B3)                               \
    {                                                                                 \
        v8h ha0 = {(_Float16)A0.x, (_Float16)A0.y, (_Float16)A0.z, (_Float16)A0.w,    \
                   (_Float16)A1.x, (_Float16)A1.y, (_Float16)A1.z, (_Float16)A1.w};   \
        v8h ha1 = {(_Float16)A2.x, (_Float16)A2.y, (_Float16)A2.z, (_Float16)A2.w,    \
                   (_Float16)A3.x, (_Float16)A3.y, (_Float16)A3.z, (_Float16)A3.w};   \
        v8h hb0 = {(_Float16)B0.x, (_Float16)B0.y, (_Float16)B0.z, (_Float16)B0.w,    \
                   (_Float16)B1.x, (_Float16)B1.y, (_Float16)B1.z, (_Float16)B1.w};   \
        v8h hb1 = {(_Float16)B2.x, (_Float16)B2.y, (_Float16)B2.z, (_Float16)B2.w,    \
                   (_Float16)B3.x, (_Float16)B3.y, (_Float16)B3.z, (_Float16)B3.w};   \
        *(v8h*)&sA[buf_][srow * LDSP + skc]     = ha0;                                \
        *(v8h*)&sA[buf_][srow * LDSP + skc + 8] = ha1;                                \
        *(v8h*)&sB[buf_][srow * LDSP + skc]     = hb0;                                \
        *(v8h*)&sB[buf_][srow * LDSP + skc + 8] = hb1;                                \
    }

#define LOAD_BFRAG(dst_, buf_, nt_)                                                   \
    {                                                                                 \
        const _Float16* bpx = &sB[buf_][((nt_) * 16 + frow) * LDSP + khalfB];         \
        v8h blo = *(const v8h*)(bpx);                                                 \
        v8h bhi = *(const v8h*)(bpx + 16);                                            \
        dst_ = __builtin_shufflevector(blo, bhi, 0, 1, 2, 3, 4, 5, 6, 7,              \
                                       8, 9, 10, 11, 12, 13, 14, 15);                 \
    }

#define LOAD_AFRAG(dst_, buf_)                                                        \
    {                                                                                 \
        const _Float16* apx = &sA[buf_][(wid * 16 + frow) * LDSP + khalfA];           \
        v8h alo = *(const v8h*)(apx);                                                 \
        v8h ahi = *(const v8h*)(apx + 16);                                            \
        dst_ = __builtin_shufflevector(alo, ahi, 0, 1, 2, 3, 4, 5, 6, 7,              \
                                       8, 9, 10, 11, 12, 13, 14, 15);                 \
    }

    // ---- prologue: stage K-step 0 into buffer 0 ----
    {
        const float4* ap = (const float4*)aSrc;
        const float4* bp = (const float4*)bSrc;
        float4 a0 = ap[0], a1 = ap[1], a2 = ap[2], a3 = ap[3];
        float4 b0 = bp[0], b1 = bp[1], b2 = bp[2], b3 = bp[3];
        CVT_STORE(0, a0, a1, a2, a3, b0, b1, b2, b3);
    }
    __syncthreads();

    // ---- steady state: branchless body, last step peeled ----
#pragma unroll 2
    for (int kt = 0; kt < NK - 1; ++kt) {
        const int buf = kt & 1;

        // issue next K-step global loads first (latency hidden by WMMAs)
        const float4* ap = (const float4*)(aSrc + (kt + 1) * BK);
        const float4* bp = (const float4*)(bSrc + (kt + 1) * BK);
        float4 a0 = ap[0], a1 = ap[1], a2 = ap[2], a3 = ap[3];
        float4 b0 = bp[0], b1 = bp[1], b2 = bp[2], b3 = bp[3];

        v16h afrag;
        LOAD_AFRAG(afrag, buf);

        // group 0: B tiles 0..3
        v16h bf0[4];
#pragma unroll
        for (int nt = 0; nt < 4; ++nt) LOAD_BFRAG(bf0[nt], buf, nt);
#pragma unroll
        for (int nt = 0; nt < 4; ++nt) {
            acc[nt] = __builtin_amdgcn_wmma_f32_16x16x32_f16(
                false, afrag, false, bf0[nt], (short)0, acc[nt], false, false);
        }

        // group 1: B tiles 4..7 (independent regs -> loads overlap group-0 WMMAs)
        v16h bf1[4];
#pragma unroll
        for (int nt = 0; nt < 4; ++nt) LOAD_BFRAG(bf1[nt], buf, nt + 4);
#pragma unroll
        for (int nt = 0; nt < 4; ++nt) {
            acc[nt + 4] = __builtin_amdgcn_wmma_f32_16x16x32_f16(
                false, afrag, false, bf1[nt], (short)0, acc[nt + 4], false, false);
        }

        // convert + stage next step into the alternate buffer
        CVT_STORE(buf ^ 1, a0, a1, a2, a3, b0, b1, b2, b3);
        __syncthreads();
    }

    // ---- peeled final K-step: compute only ----
    {
        const int buf = (NK - 1) & 1;
        v16h afrag;
        LOAD_AFRAG(afrag, buf);
        v16h bf0[4];
#pragma unroll
        for (int nt = 0; nt < 4; ++nt) LOAD_BFRAG(bf0[nt], buf, nt);
#pragma unroll
        for (int nt = 0; nt < 4; ++nt) {
            acc[nt] = __builtin_amdgcn_wmma_f32_16x16x32_f16(
                false, afrag, false, bf0[nt], (short)0, acc[nt], false, false);
        }
        v16h bf1[4];
#pragma unroll
        for (int nt = 0; nt < 4; ++nt) LOAD_BFRAG(bf1[nt], buf, nt + 4);
#pragma unroll
        for (int nt = 0; nt < 4; ++nt) {
            acc[nt + 4] = __builtin_amdgcn_wmma_f32_16x16x32_f16(
                false, afrag, false, bf1[nt], (short)0, acc[nt + 4], false, false);
        }
    }

    // ---- epilogue: C layout VGPR i -> M = 8*(lane>=16)+i, N = lane&15 ----
    const int cRow = m0 + wid * 16 + (lane >> 4) * 8;
    const int cCol = n0 + (lane & 15);
#pragma unroll
    for (int nt = 0; nt < 8; ++nt) {
        float* outp = logits + (size_t)cRow * NUM_EXPERTS + cCol + nt * 16;
#pragma unroll
        for (int i = 0; i < 8; ++i) {
            outp[(size_t)i * NUM_EXPERTS] = acc[nt][i];
        }
    }
#undef CVT_STORE
#undef LOAD_BFRAG
#undef LOAD_AFRAG
}

extern "C" void kernel_launch(void* const* d_in, const int* in_sizes, int n_in,
                              void* d_out, int out_size, void* d_ws, size_t ws_size,
                              hipStream_t stream) {
    const float* x  = (const float*)d_in[0];
    const float* nw = (const float*)d_in[1];
    const float* gw = (const float*)d_in[2];

    float* normed = (float*)d_out;
    float* logits = (float*)d_out + (size_t)NUM_TOKENS * HIDDEN;

    NormGatedLinear_rmsnorm_kernel<<<NUM_TOKENS, 256, 0, stream>>>(x, nw, normed);

    dim3 grid(NUM_EXPERTS / BN, NUM_TOKENS / BM);  // 3 x 64, N fastest
    NormGatedLinear_gate_gemm_kernel<<<grid, 256, 0, stream>>>(normed, gw, logits);
}
